// Attn_40424232189956
// MI455X (gfx1250) — matrix operand order repaired and checked by reading, not compiled, not run on hardware
//
#include <hip/hip_runtime.h>

// ---------------- problem constants (match reference) ----------------
#define B_   2
#define T_   2048
#define D_   2048
#define NH_  16
#define NKV_ 4
#define HD_  128
#define BT_  (B_*T_)

typedef __attribute__((ext_vector_type(16))) __bf16 v16bf;
typedef __attribute__((ext_vector_type(8)))  float  v8f;

union FragBF {
    v16bf          v;
    unsigned int   u32[8];
    unsigned short u16[16];
};

// round-to-nearest-even f32 -> bf16
__device__ inline unsigned short f2bf(float f) {
    unsigned int u = __float_as_uint(f);
    u += 0x7FFFu + ((u >> 16) & 1u);
    return (unsigned short)(u >> 16);
}

// ---------------- f32 -> bf16 convert ----------------
__global__ void k_conv(const float* __restrict__ in, unsigned short* __restrict__ out, int n) {
    int i = blockIdx.x * blockDim.x + threadIdx.x;
    int stride = gridDim.x * blockDim.x;
    for (; i < n; i += stride) out[i] = f2bf(in[i]);
}

// A-matrix fragment (16x32 bf16), ISA 7.12.2 layout:
//   lane half kh: u32 slot i<4 -> K = 2i + 8*kh ; slot i>=4 -> K = 16 + 2(i-4) + 8*kh
//   => byte groups at {base, base+32} with base = rowBytes + 16*kh, each 16B contiguous.
// B-matrix fragment (32x16 bf16): slot i<4 -> K = 2i + 8*kh ; slot i>=4 -> K = 16 + 2(i-4) + 8*kh
//   => byte groups at {base, base+32} with base = colBytes + 16*kh, each 16B contiguous.

// ---------------- tiled WMMA GEMM: C[M,N] = X[M,K=D_] * W[N,K]^T ----------------
// block = 128 threads (4 waves); wave -> 2 row-tiles (32 rows) x 128 cols,
// every B fragment feeds two WMMAs. All hot-loop addresses are
// uniform-base + invariant u32 lane offset + immediate (GVS addressing).
// mode 0: q (rmsnorm+rope+qg, bf16 (BT,2048)) | mode 1: k (rmsnorm+rope, bf16 (BT,512))
// mode 2: v (bf16 TRANSPOSED (B,NKV*HD,T))    | mode 3: o (f32 -> d_out (BT,2048))
__global__ __launch_bounds__(128)
void k_proj(const unsigned short* __restrict__ Xb,
            const unsigned short* __restrict__ Wb,
            void* __restrict__ outp,
            const float* __restrict__ qg,
            int Nout, int mode) {
    const int lane     = threadIdx.x & 31;
    const int wave     = threadIdx.x >> 5;
    const int kh       = lane >> 4;
    const int cl       = lane & 15;
    const int rowBase0 = blockIdx.x * 128 + wave * 32;
    const int colBase  = blockIdx.y * 128;

    v8f acc[2][8];
#pragma unroll
    for (int rt = 0; rt < 2; rt++)
#pragma unroll
        for (int t = 0; t < 8; t++)
#pragma unroll
            for (int j = 0; j < 8; j++) acc[rt][t][j] = 0.f;

    const char* Xc = (const char*)Xb;
    const char* Wc = (const char*)Wb;
    const unsigned aoff0 = ((unsigned)(rowBase0 + cl)      * D_ + 8u * kh) * 2u;
    const unsigned aoff1 = ((unsigned)(rowBase0 + 16 + cl) * D_ + 8u * kh) * 2u;
    unsigned woff[8];
#pragma unroll
    for (int t = 0; t < 8; t++)
        woff[t] = ((unsigned)(colBase + t * 16 + cl) * D_ + 8u * kh) * 2u;

    for (int k0b = 0; k0b < D_ * 2; k0b += 64) {   // 32 K-elems = 64 bytes per step
        const char* xk = Xc + k0b;                 // uniform SGPR base advance
        const char* wk = Wc + k0b;
        __builtin_prefetch(xk + aoff0 + 128, 0, 1);  // global_prefetch_b8
        FragBF a0, a1;
#pragma unroll
        for (int i = 0; i < 4; i++) {
            a0.u32[i]     = *(const unsigned int*)(xk + aoff0 + 4 * i);
            a0.u32[i + 4] = *(const unsigned int*)(xk + aoff0 + 32 + 4 * i);
            a1.u32[i]     = *(const unsigned int*)(xk + aoff1 + 4 * i);
            a1.u32[i + 4] = *(const unsigned int*)(xk + aoff1 + 32 + 4 * i);
        }
#pragma unroll
        for (int t = 0; t < 8; t++) {
            FragBF bm;
#pragma unroll
            for (int i = 0; i < 8; i++)
                bm.u32[i] = *(const unsigned int*)(wk + woff[t] + 4 * i + (i & 4) * 4);
            acc[0][t] = __builtin_amdgcn_wmma_f32_16x16x32_bf16(
                false, a0.v, false, bm.v, (short)0, acc[0][t], false, false);
            acc[1][t] = __builtin_amdgcn_wmma_f32_16x16x32_bf16(
                false, a1.v, false, bm.v, (short)0, acc[1][t], false, false);
        }
    }

    if (mode == 3) {
        float* O = (float*)outp;
#pragma unroll
        for (int rt = 0; rt < 2; rt++)
#pragma unroll
            for (int t = 0; t < 8; t++)
#pragma unroll
                for (int v = 0; v < 8; v++) {
                    int row = rowBase0 + rt * 16 + v + 8 * kh;  // C layout: M = vgpr + 8*half
                    O[(size_t)row * Nout + colBase + t * 16 + cl] = acc[rt][t][v];
                }
        return;
    }

    unsigned short* O = (unsigned short*)outp;
    if (mode == 2) {
        // store V transposed: (B, NKV*HD, T) so attention reads are key-contiguous
#pragma unroll
        for (int rt = 0; rt < 2; rt++)
#pragma unroll
            for (int t = 0; t < 8; t++)
#pragma unroll
                for (int v = 0; v < 8; v++) {
                    int row  = rowBase0 + rt * 16 + v + 8 * kh;
                    int bidx = row >> 11;            // row / T_
                    int tok  = row & (T_ - 1);
                    int col  = colBase + t * 16 + cl;
                    O[((size_t)bidx * (NKV_ * HD_) + col) * T_ + tok] = f2bf(acc[rt][t][v]);
                }
        return;
    }

    // modes 0/1: RMSNorm over full head (tile width 128 == HD) then RoPE
    const int   h = colBase / HD_;
    const float g = (mode == 0) ? qg[h] : 1.0f;
    float invf[4];                                   // rope inverse freqs (per column)
#pragma unroll
    for (int t = 0; t < 4; t++) {
        int   jj = t * 16 + cl;
        float ar = (float)jj * (1.0f / 64.0f);
        float ex = powf((float)HD_ / (float)(HD_ - 2), ar);
        invf[t]  = 1.0f / (10000.0f * powf((float)T_ / 1024.0f, ex)); // T>TSL branch
    }
#pragma unroll
    for (int rt = 0; rt < 2; rt++) {
        float rms[8];
#pragma unroll
        for (int v = 0; v < 8; v++) {
            float ssq = 0.f;
#pragma unroll
            for (int t = 0; t < 8; t++) ssq += acc[rt][t][v] * acc[rt][t][v];
#pragma unroll
            for (int m = 1; m < 16; m <<= 1) ssq += __shfl_xor(ssq, m, 16);
            rms[v] = rsqrtf(ssq * (1.0f / (float)HD_) + 1.1920929e-07f);
        }
#pragma unroll
        for (int v = 0; v < 8; v++) {
            int row = rowBase0 + rt * 16 + v + 8 * kh;
            int tok = row & (T_ - 1);
#pragma unroll
            for (int t = 0; t < 4; t++) {
                float ang = (float)tok * invf[t];
                float c = cosf(ang), s = sinf(ang);
                float x1 = acc[rt][t][v] * rms[v];
                float x2 = acc[rt][t + 4][v] * rms[v];
                int   jj = t * 16 + cl;
                O[(size_t)row * Nout + colBase + jj]      = f2bf((x1 * c + x2 * s) * g);
                O[(size_t)row * Nout + colBase + 64 + jj] = f2bf((-x1 * s + x2 * c) * g);
            }
        }
    }
}

// ---------------- flash attention: 4 independent waves, 16 queries each ----------------
__global__ __launch_bounds__(128)
void k_attn(const unsigned short* __restrict__ Qb,   // (B*T, 2048) bf16 post-rope
            const unsigned short* __restrict__ Kb,   // (B*T, 512)  bf16 post-rope
            const unsigned short* __restrict__ Vt,   // (B, NKV*HD, T) bf16 transposed
            unsigned short* __restrict__ Yb) {       // (B*T, 2048) bf16
    __shared__ unsigned short sP[4 * 16 * 32];       // wave-private P tiles

    const int lane  = threadIdx.x & 31;
    const int wave  = threadIdx.x >> 5;
    const int kh    = lane >> 4;
    const int cl    = lane & 15;
    const int h     = blockIdx.y;
    const int b     = blockIdx.z;
    const int kvh   = h / (NH_ / NKV_);
    const int qBase = blockIdx.x * 64 + wave * 16;

    const unsigned kvS = (unsigned)(NKV_ * HD_);     // 512
    const char* Kc = (const char*)(Kb + ((size_t)b * T_) * kvS + (size_t)kvh * HD_);
    const char* Vc = (const char*)(Vt + ((size_t)b * (NKV_ * HD_) + (size_t)kvh * HD_) * T_);

    // Q held as 4 A-fragments (16x32 each) covering HD=128
    FragBF aq[4];
    {
        const char* qrow = (const char*)(Qb + ((size_t)b * T_ + qBase + cl) * D_ + (size_t)h * HD_);
        const unsigned qo = 16u * kh;                // bytes
#pragma unroll
        for (int s = 0; s < 4; s++)
#pragma unroll
            for (int i = 0; i < 4; i++) {
                aq[s].u32[i]     = *(const unsigned int*)(qrow + qo + s * 64 + 4 * i);
                aq[s].u32[i + 4] = *(const unsigned int*)(qrow + qo + s * 64 + 32 + 4 * i);
            }
    }

    v8f o[8];
#pragma unroll
    for (int t = 0; t < 8; t++)
#pragma unroll
        for (int j = 0; j < 8; j++) o[t][j] = 0.f;
    float mrow[8], lrow[8];
#pragma unroll
    for (int v = 0; v < 8; v++) { mrow[v] = -__builtin_inff(); lrow[v] = 0.f; }

    // invariant per-lane byte offsets (GVS addressing in the KV loop)
    unsigned koff[2];
    koff[0] = ((unsigned)cl        * kvS + 8u * kh) * 2u;
    koff[1] = ((unsigned)(16 + cl) * kvS + 8u * kh) * 2u;
    unsigned voff[8];
#pragma unroll
    for (int dt = 0; dt < 8; dt++)
        voff[dt] = ((unsigned)(dt * 16 + cl) * T_ + 8u * kh) * 2u;

    const float scale = 0.08838834764831845f;        // 1/sqrt(128)
    const int   nkb   = (qBase + 15) / 32 + 1;       // causal block bound
    unsigned short* myP = sP + wave * 512;

    for (int kb = 0; kb < nkb; ++kb) {
        const int kBase = kb * 32;
        const char* kk = Kc + (size_t)kBase * (kvS * 2);  // uniform SALU advance
        const char* vk = Vc + (size_t)kBase * 2;

        // prefetch next block's K rows / Vt columns
        __builtin_prefetch(kk + (32u + lane) * (kvS * 2), 0, 1);
        __builtin_prefetch(vk + (unsigned)lane        * (T_ * 2) + 128, 0, 1);
        __builtin_prefetch(vk + (unsigned)(lane + 64) * (T_ * 2) + 128, 0, 1);

        // scores S = Q (16x128) * K^T (128x32), two 16-key n-tiles
        v8f sc[2];
#pragma unroll
        for (int nt = 0; nt < 2; nt++) {
            v8f s;
#pragma unroll
            for (int j = 0; j < 8; j++) s[j] = 0.f;
#pragma unroll
            for (int sl = 0; sl < 4; sl++) {
                FragBF bk;
#pragma unroll
                for (int i = 0; i < 8; i++)
                    bk.u32[i] = *(const unsigned int*)(kk + koff[nt] + sl * 64 + 4 * i + (i & 4) * 4);
                s = __builtin_amdgcn_wmma_f32_16x16x32_bf16(
                    false, aq[sl].v, false, bk.v, (short)0, s, false, false);
            }
            sc[nt] = s;
        }

        // online softmax (row stats replicated across each lane-half)
        float p0s[8], p1s[8];
#pragma unroll
        for (int v = 0; v < 8; v++) {
            const int query = qBase + v + 8 * kh;
            float s0 = (kBase + cl      <= query) ? sc[0][v] * scale : -__builtin_inff();
            float s1 = (kBase + 16 + cl <= query) ? sc[1][v] * scale : -__builtin_inff();
            float mx = fmaxf(s0, s1);
#pragma unroll
            for (int m = 1; m < 16; m <<= 1) mx = fmaxf(mx, __shfl_xor(mx, m, 16));
            float mnew = fmaxf(mrow[v], mx);
            float corr = expf(mrow[v] - mnew);
            float p0 = expf(s0 - mnew), p1 = expf(s1 - mnew);
            float ps = p0 + p1;
#pragma unroll
            for (int m = 1; m < 16; m <<= 1) ps += __shfl_xor(ps, m, 16);
            lrow[v] = lrow[v] * corr + ps;
            mrow[v] = mnew;
#pragma unroll
            for (int t = 0; t < 8; t++) o[t][v] *= corr;
            p0s[v] = p0; p1s[v] = p1;
        }

        // transpose P (C layout -> A layout) through wave-private LDS (in-order per wave)
#pragma unroll
        for (int v = 0; v < 8; v++) {
            int M = v + 8 * kh;
            myP[M * 32 + cl]      = f2bf(p0s[v]);
            myP[M * 32 + 16 + cl] = f2bf(p1s[v]);
        }
        FragBF pa;
        {
            const char* pr = (const char*)(myP + cl * 32) + 16u * kh;
#pragma unroll
            for (int i = 0; i < 4; i++) {
                pa.u32[i]     = *(const unsigned int*)(pr + 4 * i);
                pa.u32[i + 4] = *(const unsigned int*)(pr + 32 + 4 * i);
            }
        }

        // O += P (16x32) * V (32x128); Vt makes B-fragment reads key-contiguous
#pragma unroll
        for (int dt = 0; dt < 8; dt++) {
            FragBF bv;
#pragma unroll
            for (int i = 0; i < 8; i++)
                bv.u32[i] = *(const unsigned int*)(vk + voff[dt] + 4 * i + (i & 4) * 4);
            o[dt] = __builtin_amdgcn_wmma_f32_16x16x32_bf16(
                false, pa.v, false, bv.v, (short)0, o[dt], false, false);
        }
    }

    // normalize and store y (bf16) at (b, tok, h*128 + d)
#pragma unroll
    for (int v = 0; v < 8; v++) {
        int tok = qBase + v + 8 * kh;
        float linv = 1.0f / lrow[v];
        size_t rowoff = ((size_t)b * T_ + tok) * D_ + (size_t)h * HD_;
#pragma unroll
        for (int dt = 0; dt < 8; dt++)
            Yb[rowoff + dt * 16 + cl] = f2bf(o[dt][v] * linv);
    }
}

// ---------------- launch ----------------
extern "C" void kernel_launch(void* const* d_in, const int* in_sizes, int n_in,
                              void* d_out, int out_size, void* d_ws, size_t ws_size,
                              hipStream_t stream) {
    (void)in_sizes; (void)n_in; (void)out_size; (void)ws_size;
    const float* x  = (const float*)d_in[0];
    const float* qw = (const float*)d_in[1];
    const float* kw = (const float*)d_in[2];
    const float* vw = (const float*)d_in[3];
    const float* ow = (const float*)d_in[4];
    const float* qg = (const float*)d_in[5];

    char* ws = (char*)d_ws;
    size_t off = 0;
    auto alloc = [&](size_t elems) -> unsigned short* {
        unsigned short* p = (unsigned short*)(ws + off);
        off += (elems * sizeof(unsigned short) + 255) & ~(size_t)255;
        return p;
    };
    unsigned short* xb  = alloc((size_t)BT_ * D_);
    unsigned short* qwb = alloc((size_t)D_ * D_);
    unsigned short* kwb = alloc((size_t)NKV_ * HD_ * D_);
    unsigned short* vwb = alloc((size_t)NKV_ * HD_ * D_);
    unsigned short* owb = alloc((size_t)D_ * D_);
    unsigned short* qr  = alloc((size_t)BT_ * D_);
    unsigned short* kr  = alloc((size_t)BT_ * NKV_ * HD_);
    unsigned short* vtb = alloc((size_t)BT_ * NKV_ * HD_);   // transposed V
    unsigned short* yb  = alloc((size_t)BT_ * D_);

    k_conv<<<dim3(2048), 256, 0, stream>>>(x,  xb,  BT_ * D_);
    k_conv<<<dim3(1024), 256, 0, stream>>>(qw, qwb, D_ * D_);
    k_conv<<<dim3(256),  256, 0, stream>>>(kw, kwb, NKV_ * HD_ * D_);
    k_conv<<<dim3(256),  256, 0, stream>>>(vw, vwb, NKV_ * HD_ * D_);
    k_conv<<<dim3(1024), 256, 0, stream>>>(ow, owb, D_ * D_);

    k_proj<<<dim3(BT_ / 128, D_ / 128),           128, 0, stream>>>(xb, qwb, qr,  qg, D_,         0);
    k_proj<<<dim3(BT_ / 128, (NKV_ * HD_) / 128), 128, 0, stream>>>(xb, kwb, kr,  qg, NKV_ * HD_, 1);
    k_proj<<<dim3(BT_ / 128, (NKV_ * HD_) / 128), 128, 0, stream>>>(xb, vwb, vtb, qg, NKV_ * HD_, 2);

    k_attn<<<dim3(T_ / 64, NH_, B_), 128, 0, stream>>>(qr, kr, vtb, yb);

    k_proj<<<dim3(BT_ / 128, D_ / 128), 128, 0, stream>>>(yb, owb, d_out, qg, D_, 3);
}
